// DynamicDomainFilters_19542101196961
// MI455X (gfx1250) — matrix operand order repaired and checked by reading, not compiled, not run on hardware
//
#include <hip/hip_runtime.h>

#define N_S    32
#define C_IN   128
#define CM     32
#define H_     96
#define W_     96
#define HW     9216            // 96*96
#define PLANES 128
#define NGROUP (N_S * HW / 16) // 18432 wave-sized pixel groups

typedef __attribute__((ext_vector_type(16))) __bf16 v16bf;
typedef __attribute__((ext_vector_type(8)))  float  v8f;
typedef __attribute__((ext_vector_type(2)))  float  v2f;

// ---------------------------------------------------------------------------
// Kernel 1: adaptive 3x3 max pool (32x32 block max). One workgroup per
// (n, c, a, b) block. Also serves to pull x into the 192MB L2 ahead of the
// WMMA pass.
// ---------------------------------------------------------------------------
__global__ __launch_bounds__(256) void pool_max_kernel(
    const float* __restrict__ x, float* __restrict__ pooled) {
  __shared__ float red[256];
  int b  = blockIdx.x;                 // b = ((n*C + c)*9 + ab)
  int ab = b % 9;
  int c  = (b / 9) % C_IN;
  int n  = b / (9 * C_IN);
  int a  = ab / 3, bb = ab % 3;
  const float* xp = x + (size_t)(n * C_IN + c) * HW + (a * 32) * W_ + bb * 32;
  int t = threadIdx.x;
  float m = -3.402823e38f;
  for (int e = t; e < 1024; e += 256) {
    int r = e >> 5, col = e & 31;
    m = fmaxf(m, xp[r * W_ + col]);
  }
  red[t] = m;
  __syncthreads();
  for (int s = 128; s > 0; s >>= 1) {
    if (t < s) red[t] = fmaxf(red[t], red[t + s]);
    __syncthreads();
  }
  if (t == 0) pooled[b] = red[0];
}

// ---------------------------------------------------------------------------
// Kernel 2: per-sample  g = relu(W1*pooled + b1)  then  Gw = w_fuse * g.
// Tiny (one workgroup per sample).
// ---------------------------------------------------------------------------
__global__ __launch_bounds__(256) void gmat_kernel(
    const float* __restrict__ pooled, const float* __restrict__ w1,
    const float* __restrict__ b1, const float* __restrict__ w_fuse,
    float* __restrict__ Gw) {
  __shared__ float pl[C_IN * 9];
  __shared__ float gs[CM * 9];
  int n = blockIdx.x, t = threadIdx.x;
  for (int i = t; i < C_IN * 9; i += 256) pl[i] = pooled[n * C_IN * 9 + i];
  __syncthreads();
  for (int i = t; i < CM * 9; i += 256) {
    int oc = i / 9, ab = i % 9;
    float acc = b1[oc];
    for (int c = 0; c < C_IN; ++c) acc += w1[oc * C_IN + c] * pl[c * 9 + ab];
    gs[i] = fmaxf(acc, 0.f);
  }
  __syncthreads();
  for (int i = t; i < PLANES * 9; i += 256) {
    int co = i / 9, ab = i % 9;
    float acc = 0.f;
    for (int oc = 0; oc < CM; ++oc) acc += w_fuse[co * CM + oc] * gs[oc * 9 + ab];
    Gw[n * PLANES * 9 + i] = acc;
  }
}

// Kernel 2b: wsum[co] = sum_oc w_fuse, const[co] = w_fuse.b_adap + b_fuse.
__global__ void fuse_const_kernel(const float* __restrict__ w_fuse,
                                  const float* __restrict__ b_adap,
                                  const float* __restrict__ b_fuse,
                                  float* __restrict__ wsum,
                                  float* __restrict__ constv) {
  int co = threadIdx.x;
  if (co < PLANES) {
    float s = 0.f, cv = 0.f;
    for (int oc = 0; oc < CM; ++oc) {
      float w = w_fuse[co * CM + oc];
      s += w;
      cv += w * b_adap[oc];
    }
    wsum[co]   = s;
    constv[co] = cv + b_fuse[co];
  }
}

// ---------------------------------------------------------------------------
// Kernel 3: main GEMM  f = relu(W1 x + b1)  via v_wmma_f32_16x16x32_bf16,
// fused with channel contractions s1 = wk.f, s2 = bk.f (f never stored).
// One wave handles 16 pixels: M=32 (2 m-tiles), K=128 (4 k-steps) => 8 WMMA.
// ---------------------------------------------------------------------------
__global__ __launch_bounds__(256) void f_reduce_kernel(
    const float* __restrict__ x, const float* __restrict__ w1,
    const float* __restrict__ b1, const float* __restrict__ wk,
    const float* __restrict__ bk, float* __restrict__ s1,
    float* __restrict__ s2) {
  __shared__ float wk_s[CM], bk_s[CM], b1_s[CM];
  int t = threadIdx.x;
  if (t < CM) { wk_s[t] = wk[t]; bk_s[t] = bk[t]; b1_s[t] = b1[t]; }
  __syncthreads();

  int lane = t & 31, wave = t >> 5;
  int gid  = blockIdx.x * 8 + wave;          // 0..NGROUP-1
  int n    = gid / (HW / 16);
  int p0   = (gid % (HW / 16)) * 16;
  int half = lane >> 4, m = lane & 15;
  int p    = p0 + m;

  // A-operand: W1 tiles in bf16, ISA 16-bit 16x32 A layout
  // (VGPR v: K = 2v..2v+1 for v<4, 16+2(v-4).. for v>=4; +8 for upper half).
  v16bf A[2][4];
  for (int mt = 0; mt < 2; ++mt)
    for (int kt = 0; kt < 4; ++kt) {
      int oc = mt * 16 + m;
      for (int j = 0; j < 16; ++j) {
        int v    = j >> 1;
        int base = (v < 4 ? 2 * v : 16 + 2 * (v - 4)) + 8 * half;
        int k    = kt * 32 + base + (j & 1);
        A[mt][kt][j] = (__bf16)w1[oc * C_IN + k];
      }
    }

  const float* xn = x + (size_t)n * C_IN * HW;
  __builtin_prefetch(xn + p, 0, 0);          // global_prefetch_b8

  v8f acc0 = {}, acc1 = {};
  for (int kt = 0; kt < 4; ++kt) {
    // B-operand 32x16 (KxN) bf16: lanes 0-15 hold K=0..7,16..23 of column N=lane,
    // lanes 16-31 hold K=8..15,24..31. Coalesced 64B per channel row per half-wave.
    v16bf B;
    for (int j = 0; j < 16; ++j) {
      int k = kt * 32 + (j < 8 ? 8 * half + j : 16 + 8 * half + (j - 8));
      B[j] = (__bf16)xn[(size_t)k * HW + p];
    }
    acc0 = __builtin_amdgcn_wmma_f32_16x16x32_bf16(false, A[0][kt], false, B,
                                                   (short)0, acc0, false, false);
    acc1 = __builtin_amdgcn_wmma_f32_16x16x32_bf16(false, A[1][kt], false, B,
                                                   (short)0, acc1, false, false);
  }

  // relu + weighted channel reduction on the f32 accumulators
  // (C/D layout: VGPR r, lane l -> M = r + 8*(l/16), N = l%16).
  float p1 = 0.f, p2 = 0.f;
  for (int r = 0; r < 8; ++r) {
    int oc0  = r + 8 * half;
    float v0 = fmaxf(acc0[r] + b1_s[oc0], 0.f);
    p1 += wk_s[oc0] * v0;
    p2 += bk_s[oc0] * v0;
    int oc1  = 16 + oc0;
    float v1 = fmaxf(acc1[r] + b1_s[oc1], 0.f);
    p1 += wk_s[oc1] * v1;
    p2 += bk_s[oc1] * v1;
  }
  // combine the two M-halves that share a pixel column
  p1 += __shfl_xor(p1, 16, 32);
  p2 += __shfl_xor(p2, 16, 32);
  if (lane < 16) {
    s1[(size_t)n * HW + p] = p1;
    s2[(size_t)n * HW + p] = p2;
  }
}

// ---------------------------------------------------------------------------
// Kernel 4: out[n,co,p] = sum_ab Gw[n,co,ab]*s1[p+off_ab]
//                         + wsum[co]*box3x3(s2)[p] + const[co]
// as a K=12 GEMM (rows 0-8: shifted s1, row 9: box(s2), rows 10-11: zero)
// via v_wmma_f32_16x16x4_f32: 8 m-tiles x 3 k-steps per 16 pixels.
// ---------------------------------------------------------------------------
__global__ __launch_bounds__(256) void out_kernel(
    const float* __restrict__ s1, const float* __restrict__ s2,
    const float* __restrict__ Gw, const float* __restrict__ wsum,
    const float* __restrict__ constv, float* __restrict__ out) {
  int t = threadIdx.x, lane = t & 31, wave = t >> 5;
  int gid  = blockIdx.x * 8 + wave;
  int n    = gid / (HW / 16);
  int p0   = (gid % (HW / 16)) * 16;       // 16 pixels, same image row
  int half = lane >> 4, m = lane & 15;
  int h    = p0 / W_, w = (p0 % W_) + m;

  const float* s1n = s1 + (size_t)n * HW;
  const float* s2n = s2 + (size_t)n * HW;

  // B-operand 4x16 f32: lane = column (pixel), K = 2*half + j per k-step.
  v2f B[3];
  for (int kt = 0; kt < 3; ++kt)
    for (int j = 0; j < 2; ++j) {
      int k   = kt * 4 + 2 * half + j;
      float v = 0.f;
      if (k < 9) {
        int hh = h + k / 3 - 1, ww = w + k % 3 - 1;
        if (hh >= 0 && hh < H_ && ww >= 0 && ww < W_) v = s1n[hh * W_ + ww];
      } else if (k == 9) {
        for (int dh = -1; dh <= 1; ++dh)
          for (int dw = -1; dw <= 1; ++dw) {
            int hh = h + dh, ww = w + dw;
            if (hh >= 0 && hh < H_ && ww >= 0 && ww < W_) v += s2n[hh * W_ + ww];
          }
      }
      B[kt][j] = v;
    }

  const float* Gn = Gw + (size_t)n * PLANES * 9;
  for (int mt = 0; mt < 8; ++mt) {
    int row = mt * 16 + m;                 // A row (co), M = lane%16
    v2f A[3];
    for (int kt = 0; kt < 3; ++kt)
      for (int j = 0; j < 2; ++j) {
        int k   = kt * 4 + 2 * half + j;
        float v = 0.f;
        if (k < 9)       v = Gn[row * 9 + k];
        else if (k == 9) v = wsum[row];
        A[kt][j] = v;
      }
    v8f acc = {};
    for (int kt = 0; kt < 3; ++kt)
      acc = __builtin_amdgcn_wmma_f32_16x16x4_f32(false, A[kt], false, B[kt],
                                                  (short)0, acc, false, false);
    for (int r = 0; r < 8; ++r) {
      int co = mt * 16 + r + 8 * half;
      out[(size_t)(n * PLANES + co) * HW + p0 + m] = acc[r] + constv[co];
    }
  }
}

// ---------------------------------------------------------------------------
extern "C" void kernel_launch(void* const* d_in, const int* in_sizes, int n_in,
                              void* d_out, int out_size, void* d_ws,
                              size_t ws_size, hipStream_t stream) {
  const float* x      = (const float*)d_in[0];
  const float* w1     = (const float*)d_in[1];
  const float* b1     = (const float*)d_in[2];
  const float* wk     = (const float*)d_in[3];
  const float* bk     = (const float*)d_in[4];
  const float* b_adap = (const float*)d_in[5];
  const float* w_fuse = (const float*)d_in[6];
  const float* b_fuse = (const float*)d_in[7];
  float* out = (float*)d_out;

  // workspace layout (floats): pooled | Gw | wsum | const | s1 | s2  (~2.7 MB)
  float* ws     = (float*)d_ws;
  float* pooled = ws;                 // 32*128*9   = 36864
  float* Gw     = ws + 36864;         // 32*128*9   = 36864
  float* wsum   = ws + 73728;         // 128
  float* constv = ws + 73856;         // 128
  float* s1     = ws + 73984;         // 32*9216    = 294912
  float* s2     = ws + 368896;        // 32*9216    = 294912

  // pooling first: also primes the 192MB L2 with x for the WMMA pass
  hipLaunchKernelGGL(pool_max_kernel, dim3(N_S * C_IN * 9), dim3(256), 0,
                     stream, x, pooled);
  hipLaunchKernelGGL(gmat_kernel, dim3(N_S), dim3(256), 0, stream, pooled, w1,
                     b1, w_fuse, Gw);
  hipLaunchKernelGGL(fuse_const_kernel, dim3(1), dim3(128), 0, stream, w_fuse,
                     b_adap, b_fuse, wsum, constv);
  hipLaunchKernelGGL(f_reduce_kernel, dim3(NGROUP / 8), dim3(256), 0, stream,
                     x, w1, b1, wk, bk, s1, s2);
  hipLaunchKernelGGL(out_kernel, dim3(NGROUP / 8), dim3(256), 0, stream, s1,
                     s2, Gw, wsum, constv, out);
}
